// LearnableFiltration_23313082483163
// MI455X (gfx1250) — matrix operand order repaired and checked by reading, not compiled, not run on hardware
//
#include <hip/hip_runtime.h>
#include <hip/hip_fp16.h>

typedef __attribute__((ext_vector_type(16))) _Float16 v16h;
typedef __attribute__((ext_vector_type(2)))  _Float16 h2;
typedef __attribute__((ext_vector_type(8)))  float    v8f;

#define N_PTS 1024   // N (points)
#define HDIM  32     // hidden width
#define CHUNK 16     // pairs (of 32 j-points) per wave; divides 32 => bi const

union BFrag { v16h v; h2 p[8]; };

// One wave32 owns a contiguous chunk of 16 "pairs" (32 j-points each) within a
// single (b,i) row. Layer1 is packed v_pk_fma_f16 straight into B fragments;
// layer2 is 4x v_wmma_f32_16x16x32_f16 with WMMAs and their consumers
// hand-interleaved so every WMMA -> first-D-consumer distance is filled with
// independent work (kills hazard v_nops); layer3 is lane-local relu-dot,
// merged with a single xor-16 shuffle; softplus; contiguous 128B store.
__global__ __launch_bounds__(256) void lf_mlp_wmma_kernel(
    const float* __restrict__ weights,    // [B,N]
    const float* __restrict__ distances,  // [B,N,N]
    const float* __restrict__ W1,         // [3,32]
    const float* __restrict__ b1,         // [32]
    const float* __restrict__ W2,         // [32,32] (h_in, k_out)
    const float* __restrict__ b2,         // [32]
    const float* __restrict__ W3,         // [32,1]
    const float* __restrict__ b3,         // [1]
    float* __restrict__ out,              // [B,N,N]
    int numPairs)                         // B*N*N / 32
{
    const int lane     = threadIdx.x & 31;
    const int laneHi   = lane >> 4;     // 0: low half, 1: high half
    const int col      = lane & 15;     // this lane's point within a sub-tile
    const bool lowHalf = lane < 16;

    const int wave = blockIdx.x * (blockDim.x >> 5) + (threadIdx.x >> 5);
    const int tp0  = wave * CHUNK;
    if (tp0 >= numPairs) return;
    const int tpEnd = (tp0 + CHUNK < numPairs) ? tp0 + CHUNK : numPairs;

    // ---- loop-invariant per-lane constants ------------------------------
    // Layer-1 params, packed f16 pairs, in B-fragment K-slot order.
    // B (16-bit, 32x16): lane n holds col (point) n; this code's order (A matches): slot s -> K = s + laneHi*16
    h2 w1p1[8], w1p2[8], prep[8];
    {
        // bi = b*N + i is constant for the whole chunk (CHUNK divides 32)
        const float wi = weights[tp0 >> 5];
#pragma unroll
        for (int p = 0; p < 8; ++p) {
            int k = 2 * p + (laneHi << 4);
            w1p1[p] = (h2){(_Float16)W1[1 * HDIM + k], (_Float16)W1[1 * HDIM + k + 1]};
            w1p2[p] = (h2){(_Float16)W1[2 * HDIM + k], (_Float16)W1[2 * HDIM + k + 1]};
            prep[p] = (h2){(_Float16)fmaf(wi, W1[0 * HDIM + k],     b1[k]),
                           (_Float16)fmaf(wi, W1[0 * HDIM + k + 1], b1[k + 1])};
        }
    }
    // W2^T as two A fragments: A[m][k] = W2[k][m]  (m = output feature)
    // A (16-bit, 16x32): lane L holds row M=L&15; this code's order (same as B): slot s -> K = s+laneHi*16
    v16h a0, a1;
#pragma unroll
    for (int s = 0; s < 16; ++s) {
        int k = s + (laneHi << 4);
        a0[s] = (_Float16)W2[k * HDIM + col];        // feats 0-15
        a1[s] = (_Float16)W2[k * HDIM + 16 + col];   // feats 16-31
    }
    // Bias baked into C: C[m][n] = b2[m]. C VGPR r: low lanes feat r (+0/16),
    // high lanes feat r+8 (+0/16). Same indexing for the W3 coefficients.
    v8f cin0, cin1;
    float w3c0[8], w3c1[8];
#pragma unroll
    for (int r = 0; r < 8; ++r) {
        int f = r + (laneHi << 3);
        cin0[r] = b2[f];
        cin1[r] = b2[16 + f];
        w3c0[r] = W3[f];
        w3c1[r] = W3[16 + f];
    }
    const float b3v = b3[0];
    const h2 zero2 = (h2){(_Float16)0.0f, (_Float16)0.0f};

    const int bi   = tp0 >> 5;
    const int base = bi << 10;                  // row offset into [B*N, N]
    const int bRow = bi & ~(N_PTS - 1);         // b*N for weights[b, j]

    for (int tp = tp0; tp < tpEnd; ++tp) {
        const int j0 = (tp & 31) << 5;          // 32 consecutive j per pair

        // prefetch next iteration's 128B distance block (wave-uniform branch)
        if (tp + 1 < tpEnd) {
            __builtin_prefetch(
                distances + base + (((tp + 1) & 31) << 5) + lane, 0, 1);
        }

        const int jc0 = j0 + col;
        const int jc1 = j0 + 16 + col;
        const float wj0 = weights[bRow + jc0];
        const float wj1 = weights[bRow + jc1];
        const float dd0 = distances[base + jc0];
        const float dd1 = distances[base + jc1];

        const _Float16 wj0h = (_Float16)wj0, wj1h = (_Float16)wj1;
        const _Float16 dd0h = (_Float16)dd0, dd1h = (_Float16)dd1;
        const h2 wj0v = (h2){wj0h, wj0h}, wj1v = (h2){wj1h, wj1h};
        const h2 dd0v = (h2){dd0h, dd0h}, dd1v = (h2){dd1h, dd1h};

        // ---- layer 1 (sub-tile 0) -> bf0 --------------------------------
        BFrag bf0;
#pragma unroll
        for (int p = 0; p < 8; ++p) {
            h2 t0 = __builtin_elementwise_fma(dd0v, w1p2[p], prep[p]);
            h2 h0 = __builtin_elementwise_fma(wj0v, w1p1[p], t0);
            bf0.p[p] = __builtin_elementwise_max(h0, zero2);
        }

        v8f c00 = __builtin_amdgcn_wmma_f32_16x16x32_f16(
            false, a0, false, bf0.v, (short)0, cin0, false, false);

        // ---- layer 1 (sub-tile 1) -> bf1  (fills c00's hazard window) ---
        BFrag bf1;
#pragma unroll
        for (int p = 0; p < 8; ++p) {
            h2 t1 = __builtin_elementwise_fma(dd1v, w1p2[p], prep[p]);
            h2 h1 = __builtin_elementwise_fma(wj1v, w1p1[p], t1);
            bf1.p[p] = __builtin_elementwise_max(h1, zero2);
        }

        v8f c01 = __builtin_amdgcn_wmma_f32_16x16x32_f16(
            false, a1, false, bf0.v, (short)0, cin1, false, false);

        // consume c00 (fills c01's hazard window)
        float acc0 = 0.0f;
#pragma unroll
        for (int r = 0; r < 8; ++r)
            acc0 = fmaf(fmaxf(c00[r], 0.0f), w3c0[r], acc0);

        v8f c10 = __builtin_amdgcn_wmma_f32_16x16x32_f16(
            false, a0, false, bf1.v, (short)0, cin0, false, false);

        // consume c01 (fills c10's hazard window)
#pragma unroll
        for (int r = 0; r < 8; ++r)
            acc0 = fmaf(fmaxf(c01[r], 0.0f), w3c1[r], acc0);

        v8f c11 = __builtin_amdgcn_wmma_f32_16x16x32_f16(
            false, a1, false, bf1.v, (short)0, cin1, false, false);

        // consume c10 (fills c11's hazard window)
        float acc1 = 0.0f;
#pragma unroll
        for (int r = 0; r < 8; ++r)
            acc1 = fmaf(fmaxf(c10[r], 0.0f), w3c0[r], acc1);

        // merge setup (independent of c11): pick local keep/send halves
        const float keepA = lowHalf ? acc0 : 0.0f;   // partial; acc1 added below

        // consume c11
#pragma unroll
        for (int r = 0; r < 8; ++r)
            acc1 = fmaf(fmaxf(c11[r], 0.0f), w3c1[r], acc1);

        // single xor-16 shuffle merge:
        //   low lanes need acc0(low)+acc0(high); high lanes acc1(high)+acc1(low)
        const float keep = lowHalf ? acc0 : acc1;   // local half
        const float send = lowHalf ? acc1 : acc0;   // remote half's partner
        const float res  = keep + __shfl_xor(send, 16);
        (void)keepA;

        // softplus(x) = max(x,0) + log1p(exp(-|x|))  (stable)
        const float x  = res + b3v;
        const float sp = fmaxf(x, 0.0f) + log1pf(__expf(-fabsf(x)));

        // one contiguous 128B store across the whole wave
        out[base + j0 + lane] = sp;
    }
}

extern "C" void kernel_launch(void* const* d_in, const int* in_sizes, int n_in,
                              void* d_out, int out_size, void* d_ws, size_t ws_size,
                              hipStream_t stream) {
    const float* weights   = (const float*)d_in[0];
    const float* distances = (const float*)d_in[1];
    const float* W1 = (const float*)d_in[2];
    const float* b1 = (const float*)d_in[3];
    const float* W2 = (const float*)d_in[4];
    const float* b2 = (const float*)d_in[5];
    const float* W3 = (const float*)d_in[6];
    const float* b3 = (const float*)d_in[7];
    float* out = (float*)d_out;

    const int numPairs = out_size / 32;                     // B*N*N / 32
    const int wavesNeeded = (numPairs + CHUNK - 1) / CHUNK; // 1 chunk per wave
    const int wavesPerBlock = 8;                            // 256 threads
    int blocks = (wavesNeeded + wavesPerBlock - 1) / wavesPerBlock;
    if (blocks < 1) blocks = 1;

    lf_mlp_wmma_kernel<<<blocks, 256, 0, stream>>>(
        weights, distances, W1, b1, W2, b2, W3, b3, out, numPairs);
}
